// GPT2SelfAttention_1700807049335
// MI455X (gfx1250) — matrix operand order repaired and checked by reading, not compiled, not run on hardware
//
#include <hip/hip_runtime.h>
#include <hip/hip_bf16.h>

// ---------------------------------------------------------------------------
// GPT-2 self-attention, MI455X (gfx1250, wave32).
// All four matmuls run on v_wmma_f32_16x16x32_bf16 (fp32 accumulate).
// fp32->bf16 conversion is hoisted into O(N) pre-pass kernels so the GEMM
// inner loops are pure b128 loads + WMMA (no per-element VALU conversion).
// Fragment layouts per CDNA5 ISA 7.12.2:
//   A (16x32 bf16): lane m=lane&15; VGPR0-3 K = kb + 8*(lane>>4) + 0..7,
//                   VGPR4-7 K = kb+16 + 8*(lane>>4) + 0..7   (two 16B chunks)
//   B (32x16 bf16): lane n=lane&15; K order same as A (two 16B chunks)
//   C/D (16x16 f32): elem r -> row r + 8*(lane>>4), col lane&15
// ---------------------------------------------------------------------------

typedef __attribute__((ext_vector_type(16))) __bf16 v16bf;
typedef __attribute__((ext_vector_type(8)))  float  v8f;
typedef uint4 __attribute__((may_alias)) uint4a;
typedef uint2 __attribute__((may_alias)) uint2a;

union F16x {
  v16bf v;
  uint4a q[2];
  unsigned short u[16];
};

__device__ __forceinline__ unsigned short f2bf(float f) {
  union { float f; unsigned u; } x; x.f = f;
  unsigned r = x.u + 0x7FFFu + ((x.u >> 16) & 1u);  // RNE
  return (unsigned short)(r >> 16);
}

__device__ __forceinline__ v8f wmma_bf16(const F16x& a, const F16x& b, v8f c) {
  return __builtin_amdgcn_wmma_f32_16x16x32_bf16(false, a.v, false, b.v,
                                                 (short)0, c, false, false);
}

#define B_  4
#define T_  2048
#define C_  1024
#define H_  16
#define D_  64

// ---------------------------------------------------------------------------
// Pre-pass: fp32 -> bf16 (RNE), 4 elements/thread.  Memory-bound.
// ---------------------------------------------------------------------------
__global__ __launch_bounds__(256) void cvt_bf16(const float* __restrict__ in,
                                                unsigned short* __restrict__ out,
                                                int n4) {
  const int i = blockIdx.x * 256 + threadIdx.x;
  if (i < n4) {
    const float4 f = *(const float4*)(in + (size_t)i * 4);
    uint2 o;
    o.x = (unsigned)f2bf(f.x) | ((unsigned)f2bf(f.y) << 16);
    o.y = (unsigned)f2bf(f.z) | ((unsigned)f2bf(f.w) << 16);
    *(uint2a*)(out + (size_t)i * 4) = o;
  }
}

// ---------------------------------------------------------------------------
// Shared GEMM core: acc[4][4] (64x64 wave tile), A/B both bf16, K-steps of 32.
// Per K-step: 16 b128 loads : 16 WMMAs, A and B fragments each reused 4x.
// ---------------------------------------------------------------------------
__device__ __forceinline__ void gemm_core(const unsigned short* __restrict__ A,
                                          const unsigned short* __restrict__ Bm,
                                          int Kdim, int mBase, int nBase,
                                          int lm, int hi, v8f acc[4][4]) {
#pragma unroll
  for (int i = 0; i < 4; ++i)
#pragma unroll
    for (int j = 0; j < 4; ++j) acc[i][j] = (v8f){0.f,0.f,0.f,0.f,0.f,0.f,0.f,0.f};

  for (int kk = 0; kk < Kdim; kk += 32) {
    F16x a[4], b[4];
#pragma unroll
    for (int mt = 0; mt < 4; ++mt) {
      const unsigned short* ap = A + (size_t)(mBase + mt * 16 + lm) * Kdim + kk + hi * 8;
      __builtin_prefetch(ap + 32, 0, 1);
      a[mt].q[0] = *(const uint4a*)ap;
      a[mt].q[1] = *(const uint4a*)(ap + 16);
    }
#pragma unroll
    for (int nt = 0; nt < 4; ++nt) {
      const unsigned short* bp = Bm + (size_t)(nBase + nt * 16 + lm) * Kdim + kk + hi * 8;
      __builtin_prefetch(bp + 32, 0, 1);
      b[nt].q[0] = *(const uint4a*)bp;
      b[nt].q[1] = *(const uint4a*)(bp + 16);
    }
#pragma unroll
    for (int mt = 0; mt < 4; ++mt)
#pragma unroll
      for (int nt = 0; nt < 4; ++nt) acc[mt][nt] = wmma_bf16(a[mt], b[nt], acc[mt][nt]);
  }
}

// ---------------------------------------------------------------------------
// Kernel 1: qkv = x @ w_attn^T + b_attn, scattered to Q/K [B,H,T,D] bf16 and
// V transposed [B,H,D,T] bf16.  Q pre-scaled by 1/sqrt(D).
// Block = 256 thr (8 waves, 2x4) -> 128x256 tile; wave -> 64x64.
// ---------------------------------------------------------------------------
__global__ __launch_bounds__(256) void qkv_gemm(
    const unsigned short* __restrict__ Xbf, const unsigned short* __restrict__ Wbf,
    const float* __restrict__ bias,
    unsigned short* __restrict__ Qb, unsigned short* __restrict__ Kb,
    unsigned short* __restrict__ Vtb) {
  const int lane = threadIdx.x & 31, wave = threadIdx.x >> 5;
  const int lm = lane & 15, hi = lane >> 4;
  const int mBase = blockIdx.y * 128 + (wave >> 2) * 64;
  const int nBase = blockIdx.x * 256 + (wave & 3) * 64;

  v8f acc[4][4];
  gemm_core(Xbf, Wbf, C_, mBase, nBase, lm, hi, acc);

#pragma unroll
  for (int mt = 0; mt < 4; ++mt)
#pragma unroll
    for (int nt = 0; nt < 4; ++nt)
#pragma unroll
      for (int r = 0; r < 8; ++r) {
        const int mm = mBase + mt * 16 + r + hi * 8;
        const int nn = nBase + nt * 16 + lm;
        const float vv = acc[mt][nt][r] + bias[nn];
        const int part = nn >> 10, c = nn & 1023, h = c >> 6, d = c & 63;
        const int bb = mm >> 11, t = mm & 2047, bh = bb * H_ + h;
        if (part == 0)
          Qb[((size_t)(bh * T_ + t) << 6) + d] = f2bf(vv * 0.125f);
        else if (part == 1)
          Kb[((size_t)(bh * T_ + t) << 6) + d] = f2bf(vv);
        else
          Vtb[(size_t)((bh << 6) + d) * T_ + t] = f2bf(vv);
      }
}

// ---------------------------------------------------------------------------
// Kernel 2: causal flash attention. grid = (B*H, T/256); wave = 32 q rows.
// No block-level barriers (waves have divergent k-block counts); per-wave LDS
// scratch converts P from C-layout to A-layout for the AV WMMA.
// ---------------------------------------------------------------------------
__global__ __launch_bounds__(256) void attn_kernel(
    const unsigned short* __restrict__ Qb, const unsigned short* __restrict__ Kb,
    const unsigned short* __restrict__ Vtb, unsigned short* __restrict__ AO) {
  __shared__ __align__(16) unsigned short Pbuf[8][32 * 32];
  const int lane = threadIdx.x & 31, wave = threadIdx.x >> 5;
  const int lm = lane & 15, hi = lane >> 4;
  const int bh = blockIdx.x, bb = bh >> 4, h = bh & 15;
  const int q0 = blockIdx.y * 256 + wave * 32;

  F16x qf[2][2];  // [m-tile][k-step over D]
#pragma unroll
  for (int mt = 0; mt < 2; ++mt)
#pragma unroll
    for (int ks = 0; ks < 2; ++ks) {
      const unsigned short* p =
          Qb + ((size_t)(bh * T_ + q0 + mt * 16 + lm) << 6) + ks * 32 + hi * 8;
      qf[mt][ks].q[0] = *(const uint4a*)p;
      qf[mt][ks].q[1] = *(const uint4a*)(p + 16);
    }

  v8f O[2][4];
#pragma unroll
  for (int i = 0; i < 2; ++i)
#pragma unroll
    for (int j = 0; j < 4; ++j) O[i][j] = (v8f){0.f,0.f,0.f,0.f,0.f,0.f,0.f,0.f};
  float mrun[2][8], lrun[2][8];
#pragma unroll
  for (int i = 0; i < 2; ++i)
#pragma unroll
    for (int r = 0; r < 8; ++r) { mrun[i][r] = -1e30f; lrun[i][r] = 0.f; }

  const int jd = q0 >> 5;
  for (int j = 0; j <= jd; ++j) {
    v8f S[2][2];
#pragma unroll
    for (int i = 0; i < 2; ++i)
#pragma unroll
      for (int k = 0; k < 2; ++k) S[i][k] = (v8f){0.f,0.f,0.f,0.f,0.f,0.f,0.f,0.f};

#pragma unroll
    for (int ks = 0; ks < 2; ++ks) {
      F16x kf[2];
#pragma unroll
      for (int nt = 0; nt < 2; ++nt) {
        const unsigned short* p =
            Kb + ((size_t)(bh * T_ + j * 32 + nt * 16 + lm) << 6) + ks * 32 + hi * 8;
        kf[nt].q[0] = *(const uint4a*)p;
        kf[nt].q[1] = *(const uint4a*)(p + 16);
      }
#pragma unroll
      for (int mt = 0; mt < 2; ++mt)
#pragma unroll
        for (int nt = 0; nt < 2; ++nt) S[mt][nt] = wmma_bf16(qf[mt][ks], kf[nt], S[mt][nt]);
    }

    if (j == jd) {  // triangular mask only on the diagonal block
#pragma unroll
      for (int mt = 0; mt < 2; ++mt)
#pragma unroll
        for (int nt = 0; nt < 2; ++nt)
#pragma unroll
          for (int r = 0; r < 8; ++r) {
            const int qg = q0 + mt * 16 + r + hi * 8;
            const int kg = j * 32 + nt * 16 + lm;
            if (kg > qg) S[mt][nt][r] = -1e30f;
          }
    }

#pragma unroll
    for (int mt = 0; mt < 2; ++mt)
#pragma unroll
      for (int r = 0; r < 8; ++r) {
        float rv = fmaxf(S[mt][0][r], S[mt][1][r]);
#pragma unroll
        for (int off = 8; off >= 1; off >>= 1) rv = fmaxf(rv, __shfl_xor(rv, off, 32));
        const float mo = mrun[mt][r];
        const float mn = fmaxf(mo, rv);
        const float al = __expf(mo - mn);
        const float p0 = __expf(S[mt][0][r] - mn);
        const float p1 = __expf(S[mt][1][r] - mn);
        S[mt][0][r] = p0; S[mt][1][r] = p1;
        float rs = p0 + p1;
#pragma unroll
        for (int off = 8; off >= 1; off >>= 1) rs += __shfl_xor(rs, off, 32);
        lrun[mt][r] = lrun[mt][r] * al + rs;
        mrun[mt][r] = mn;
#pragma unroll
        for (int nt4 = 0; nt4 < 4; ++nt4) O[mt][nt4][r] *= al;
      }

    // P: C-layout -> LDS -> A-layout (bf16)
    unsigned short* pb = Pbuf[wave];
#pragma unroll
    for (int mt = 0; mt < 2; ++mt)
#pragma unroll
      for (int nt = 0; nt < 2; ++nt)
#pragma unroll
        for (int r = 0; r < 8; ++r)
          pb[(mt * 16 + r + hi * 8) * 32 + nt * 16 + lm] = f2bf(S[mt][nt][r]);

    F16x pa[2];
#pragma unroll
    for (int mt = 0; mt < 2; ++mt) {
      const unsigned short* pp = pb + (mt * 16 + lm) * 32 + hi * 8;
      pa[mt].q[0] = *(const uint4a*)pp;
      pa[mt].q[1] = *(const uint4a*)(pp + 16);
    }

    F16x vf[4];
#pragma unroll
    for (int nt4 = 0; nt4 < 4; ++nt4) {
      const unsigned short* vp =
          Vtb + (size_t)((bh << 6) + nt4 * 16 + lm) * T_ + j * 32 + hi * 8;
      vf[nt4].q[0] = *(const uint4a*)vp;
      vf[nt4].q[1] = *(const uint4a*)(vp + 16);
    }
#pragma unroll
    for (int mt = 0; mt < 2; ++mt)
#pragma unroll
      for (int nt4 = 0; nt4 < 4; ++nt4) O[mt][nt4] = wmma_bf16(pa[mt], vf[nt4], O[mt][nt4]);
  }

  // epilogue: normalize, write attn_out bf16 [B,T,C]
#pragma unroll
  for (int mt = 0; mt < 2; ++mt)
#pragma unroll
    for (int nt4 = 0; nt4 < 4; ++nt4)
#pragma unroll
      for (int r = 0; r < 8; ++r) {
        const int t = q0 + mt * 16 + r + hi * 8;
        const int col = h * D_ + nt4 * 16 + lm;
        AO[(size_t)(bb * T_ + t) * C_ + col] = f2bf(O[mt][nt4][r] / lrun[mt][r]);
      }
}

// ---------------------------------------------------------------------------
// Kernel 3: out = attn_out @ w_proj^T + b_proj  (A and B both bf16)
// ---------------------------------------------------------------------------
__global__ __launch_bounds__(256) void proj_gemm(
    const unsigned short* __restrict__ A, const unsigned short* __restrict__ Wbf,
    const float* __restrict__ bias, float* __restrict__ Out) {
  const int lane = threadIdx.x & 31, wave = threadIdx.x >> 5;
  const int lm = lane & 15, hi = lane >> 4;
  const int mBase = blockIdx.y * 128 + (wave >> 2) * 64;
  const int nBase = blockIdx.x * 256 + (wave & 3) * 64;

  v8f acc[4][4];
  gemm_core(A, Wbf, C_, mBase, nBase, lm, hi, acc);

#pragma unroll
  for (int mt = 0; mt < 4; ++mt)
#pragma unroll
    for (int nt = 0; nt < 4; ++nt)
#pragma unroll
      for (int r = 0; r < 8; ++r) {
        const int mm = mBase + mt * 16 + r + hi * 8;
        const int nn = nBase + nt * 16 + lm;
        Out[(size_t)mm * C_ + nn] = acc[mt][nt][r] + bias[nn];
      }
}

// ---------------------------------------------------------------------------
extern "C" void kernel_launch(void* const* d_in, const int* in_sizes, int n_in,
                              void* d_out, int out_size, void* d_ws, size_t ws_size,
                              hipStream_t stream) {
  const float* x      = (const float*)d_in[0];
  const float* w_attn = (const float*)d_in[1];
  const float* b_attn = (const float*)d_in[2];
  const float* w_proj = (const float*)d_in[3];
  const float* b_proj = (const float*)d_in[4];
  float* out = (float*)d_out;

  // workspace layout (u16 elements)
  const size_t NE   = (size_t)B_ * H_ * T_ * D_;  // 8388608
  const size_t NX   = (size_t)B_ * T_ * C_;       // 8388608
  const size_t NWA  = (size_t)3 * C_ * C_;        // 3145728
  const size_t NWP  = (size_t)C_ * C_;            // 1048576
  unsigned short* ws   = (unsigned short*)d_ws;
  unsigned short* Qb   = ws;
  unsigned short* Kb   = Qb + NE;
  unsigned short* Vtb  = Kb + NE;
  unsigned short* AO   = Vtb + NE;
  unsigned short* Xbf  = AO + NX;
  unsigned short* Wabf = Xbf + NX;
  unsigned short* Wpbf = Wabf + NWA;               // total ~92 MB

  dim3 blk(256);
  // pre-convert fp32 operands to bf16 once (memory-bound)
  cvt_bf16<<<dim3((int)(NX  / 4 / 256)), blk, 0, stream>>>(x,      Xbf,  (int)(NX  / 4));
  cvt_bf16<<<dim3((int)(NWA / 4 / 256)), blk, 0, stream>>>(w_attn, Wabf, (int)(NWA / 4));
  cvt_bf16<<<dim3((int)(NWP / 4 / 256)), blk, 0, stream>>>(w_proj, Wpbf, (int)(NWP / 4));

  qkv_gemm  <<<dim3(3 * C_ / 256, B_ * T_ / 128), blk, 0, stream>>>(Xbf, Wabf, b_attn, Qb, Kb, Vtb);
  attn_kernel<<<dim3(B_ * H_, T_ / 256),          blk, 0, stream>>>(Qb, Kb, Vtb, AO);
  proj_gemm <<<dim3(C_ / 256, B_ * T_ / 128),     blk, 0, stream>>>(AO, Wpbf, b_proj, out);
}
